// DGCNN_90228672954687
// MI455X (gfx1250) — matrix operand order repaired and checked by reading, not compiled, not run on hardware
//
#include <hip/hip_runtime.h>
#include <stdint.h>

#define KNN_K 20
#define RESO 64
#define NPTS 2048
#define BATCH 4
#define BN_EPS 1e-5f

typedef __attribute__((ext_vector_type(16))) __bf16        v16bf;
typedef __attribute__((ext_vector_type(8)))  float         v8f;
typedef __attribute__((ext_vector_type(4)))  unsigned int  v4u;

union BFrag { v4u q[2]; v16bf v; };

static __device__ __forceinline__ unsigned short f2bf(float f) {
    union { float f; unsigned int u; } c; c.f = f;
    unsigned int u = c.u;
    unsigned int r = u + 0x7FFFu + ((u >> 16) & 1u);
    return (unsigned short)(r >> 16);
}
static __device__ __forceinline__ float bf2f(unsigned short h) {
    union { unsigned int u; float f; } c; c.u = ((unsigned int)h) << 16;
    return c.f;
}
static __device__ __forceinline__ void atomicMaxF(float* addr, float val) {
    int* ia = (int*)addr;
    int old = __float_as_int(*addr);
    while (__int_as_float(old) < val) {
        int assumed = old;
        old = atomicCAS(ia, assumed, __float_as_int(val));
        if (old == assumed) break;
    }
}

// ---------------------------------------------------------------- utility
__global__ void k_fill(float* p, long n, float v) {
    long i = (long)blockIdx.x * blockDim.x + threadIdx.x;
    if (i < n) p[i] = v;
}

__global__ void k_transpose_pts(const float* pts, float* x) { // [B,N,3] -> [B,3,N]
    int i = blockIdx.x * blockDim.x + threadIdx.x;
    if (i >= BATCH * NPTS) return;
    int b = i / NPTS, n = i % NPTS;
    for (int d = 0; d < 3; ++d)
        x[((size_t)b * 3 + d) * NPTS + n] = pts[((size_t)b * NPTS + n) * 3 + d];
}

// h [B][C][N] f32 -> hT [B][N][Cpad] bf16 + squared norms (of bf16-rounded values)
__global__ void k_packT(const float* h, unsigned short* hT, float* xx, int C, int Cpad) {
    int i = blockIdx.x * blockDim.x + threadIdx.x;
    if (i >= BATCH * NPTS) return;
    int b = i / NPTS, n = i % NPTS;
    float s = 0.f;
    for (int c = 0; c < Cpad; ++c) {
        float v = (c < C) ? h[((size_t)b * C + c) * NPTS + n] : 0.f;
        unsigned short bv = f2bf(v);
        hT[((size_t)b * NPTS + n) * Cpad + c] = bv;
        float vb = bf2f(bv);
        s += vb * vb;
    }
    xx[i] = s;
}

__global__ void k_packW(const float* w, unsigned short* wbf, int O, int Cf, int Cfpad) {
    int t = blockIdx.x * blockDim.x + threadIdx.x;
    if (t >= O * Cfpad) return;
    int c = t % Cfpad, o = t / Cfpad;
    wbf[t] = (c < Cf) ? f2bf(w[(size_t)o * Cf + c]) : (unsigned short)0;
}

// ---------------------------------------------------------------- kNN (gram via WMMA + top-K)
template<int NCK>
__global__ __launch_bounds__(128) void k_knn(const unsigned short* hT, const float* xx, int* idx) {
    const int Cpad = NCK * 32;
    __shared__ float dist[16][NPTS];          // 128 KB
    int b  = blockIdx.x / (NPTS / 16);
    int rt = blockIdx.x % (NPTS / 16);
    int rowbase = rt * 16;
    int tid = threadIdx.x, lane = tid & 31, wv = tid >> 5;
    int hi = lane >> 4, lm = lane & 15;
    const unsigned short* hb  = hT + (size_t)b * NPTS * Cpad;
    const float*          xxb = xx + (size_t)b * NPTS;

    BFrag A[NCK];
#pragma unroll
    for (int ch = 0; ch < NCK; ++ch) {
        const unsigned short* pa = hb + (size_t)(rowbase + lm) * Cpad + ch * 32 + hi * 8;
        A[ch].q[0] = *(const v4u*)pa;
        A[ch].q[1] = *(const v4u*)(pa + 16);
    }
    for (int ct = wv; ct < NPTS / 16; ct += 4) {
        int cb = ct * 16;
        v8f acc = {0.f,0.f,0.f,0.f,0.f,0.f,0.f,0.f};
#pragma unroll
        for (int ch = 0; ch < NCK; ++ch) {
            BFrag Bf;
            const unsigned short* pb = hb + (size_t)(cb + lm) * Cpad + ch * 32 + hi * 8;
            Bf.q[0] = *(const v4u*)pb;
            Bf.q[1] = *(const v4u*)(pb + 16);
            acc = __builtin_amdgcn_wmma_f32_16x16x32_bf16(false, A[ch].v, false, Bf.v,
                                                          (short)0, acc, false, false);
        }
        float xm = xxb[cb + lm];
#pragma unroll
        for (int i = 0; i < 8; ++i) {
            int rl = hi * 8 + i;
            dist[rl][cb + lm] = 2.f * acc[i] - xxb[rowbase + rl] - xm;
        }
    }
    __syncthreads();
    // wave wv selects top-K for rows wv*4 .. wv*4+3
    for (int r = 0; r < 4; ++r) {
        int row = wv * 4 + r;
        for (int k = 0; k < KNN_K; ++k) {
            float bv = -3.0e38f; int bc = 0;
            for (int m = lane; m < NPTS; m += 32) {
                float v = dist[row][m];
                if (v > bv) { bv = v; bc = m; }
            }
#pragma unroll
            for (int off = 16; off > 0; off >>= 1) {
                float ov = __shfl_xor(bv, off, 32);
                int   oc = __shfl_xor(bc, off, 32);
                if (ov > bv || (ov == bv && oc < bc)) { bv = ov; bc = oc; }
            }
            if (lane == 0) {
                idx[((size_t)b * NPTS + rowbase + row) * KNN_K + k] = bc;
                dist[row][bc] = -3.0e38f;
            }
        }
    }
}

// ---------------------------------------------------------------- edge-feature gather -> bf16 [col][Cfpad]
__global__ void k_gather(const float* h, const int* idx, unsigned short* f,
                         int Cin, int Cfpad, int layer0) {
    long t = (long)blockIdx.x * blockDim.x + threadIdx.x;
    long total = (long)BATCH * NPTS * KNN_K * Cfpad;
    if (t >= total) return;
    int  c   = (int)(t % Cfpad);
    long col = t / Cfpad;
    int  kk  = (int)(col % KNN_K);
    long bn  = col / KNN_K;
    int  n   = (int)(bn % NPTS);
    int  b   = (int)(bn / NPTS);
    int  j   = idx[((size_t)b * NPTS + n) * KNN_K + kk];
    float v = 0.f;
    if (layer0) {
        const float* xb = h + (size_t)b * 3 * NPTS;
        if (c < 3) v = xb[c * NPTS + j] - xb[c * NPTS + n];
        else if (c < 6) {
            float a0 = xb[0 * NPTS + j], a1 = xb[1 * NPTS + j], a2 = xb[2 * NPTS + j];
            float b0 = xb[0 * NPTS + n], b1 = xb[1 * NPTS + n], b2 = xb[2 * NPTS + n];
            int cc = c - 3;
            v = (cc == 0) ? a1 * b2 - a2 * b1 : (cc == 1) ? a2 * b0 - a0 * b2 : a0 * b1 - a1 * b0;
        } else if (c < 9) v = xb[(c - 6) * NPTS + n];
    } else {
        const float* hb = h + (size_t)b * Cin * NPTS;
        if (c < Cin)            v = hb[(size_t)c * NPTS + j] - hb[(size_t)c * NPTS + n];
        else if (c < 2 * Cin)   v = hb[(size_t)(c - Cin) * NPTS + n];
    }
    f[t] = f2bf(v);
}

// hcat [h0|h1|h2|h3] -> bf16 [col=b*N+n][512]
__global__ void k_packcat(const float* h0, const float* h1, const float* h2, const float* h3,
                          unsigned short* f) {
    long t = (long)blockIdx.x * blockDim.x + threadIdx.x;
    long total = (long)BATCH * NPTS * 512;
    if (t >= total) return;
    int c = (int)(t % 512);
    int n = (int)((t / 512) % NPTS);
    int b = (int)(t / (512L * NPTS));
    float v;
    if      (c < 64)  v = h0[((size_t)b * 64  + c)        * NPTS + n];
    else if (c < 128) v = h1[((size_t)b * 64  + (c - 64)) * NPTS + n];
    else if (c < 256) v = h2[((size_t)b * 128 + (c - 128))* NPTS + n];
    else              v = h3[((size_t)b * 256 + (c - 256))* NPTS + n];
    f[t] = f2bf(v);
}

// ---------------------------------------------------------------- conv GEMM via WMMA: pass1 (BN stats)
template<int NC>
__global__ __launch_bounds__(128) void k_conv_stats(const unsigned short* Wbf,
                                                    const unsigned short* fbf,
                                                    float* stats, int totalCols) {
    const int Cfpad = NC * 32;
    int tid = threadIdx.x, lane = tid & 31, wv = tid >> 5;
    int hi = lane >> 4, lm = lane & 15;
    int mbase = blockIdx.y * 16;
    BFrag A[NC];
#pragma unroll
    for (int ch = 0; ch < NC; ++ch) {
        const unsigned short* pa = Wbf + (size_t)(mbase + lm) * Cfpad + ch * 32 + hi * 8;
        A[ch].q[0] = *(const v4u*)pa;
        A[ch].q[1] = *(const v4u*)(pa + 16);
    }
    float s[8] = {0,0,0,0,0,0,0,0}, q[8] = {0,0,0,0,0,0,0,0};
    int T = totalCols / 16;
    int gw = blockIdx.x * 4 + wv, nw = gridDim.x * 4;
    for (int t = gw; t < T; t += nw) {
        v8f acc = {0.f,0.f,0.f,0.f,0.f,0.f,0.f,0.f};
#pragma unroll
        for (int ch = 0; ch < NC; ++ch) {
            BFrag Bf;
            const unsigned short* pb = fbf + (size_t)(t * 16 + lm) * Cfpad + ch * 32 + hi * 8;
            Bf.q[0] = *(const v4u*)pb;
            Bf.q[1] = *(const v4u*)(pb + 16);
            acc = __builtin_amdgcn_wmma_f32_16x16x32_bf16(false, A[ch].v, false, Bf.v,
                                                          (short)0, acc, false, false);
        }
#pragma unroll
        for (int i = 0; i < 8; ++i) { s[i] += acc[i]; q[i] += acc[i] * acc[i]; }
    }
#pragma unroll
    for (int i = 0; i < 8; ++i) {
        int o = mbase + hi * 8 + i;
        atomicAdd(&stats[2 * o],     s[i]);
        atomicAdd(&stats[2 * o + 1], q[i]);
    }
}

// pass2: recompute GEMM, BN + LeakyReLU, max-pool (over K or over N) via atomic max
template<int NC>
__global__ __launch_bounds__(128) void k_conv_apply(const unsigned short* Wbf,
                                                    const unsigned short* fbf,
                                                    const float* stats, const float* g, const float* bb,
                                                    float* out, int totalCols, int Kfac, int O, int Nout) {
    const int Cfpad = NC * 32;
    int tid = threadIdx.x, lane = tid & 31, wv = tid >> 5;
    int hi = lane >> 4, lm = lane & 15;
    int mbase = blockIdx.y * 16;
    BFrag A[NC];
#pragma unroll
    for (int ch = 0; ch < NC; ++ch) {
        const unsigned short* pa = Wbf + (size_t)(mbase + lm) * Cfpad + ch * 32 + hi * 8;
        A[ch].q[0] = *(const v4u*)pa;
        A[ch].q[1] = *(const v4u*)(pa + 16);
    }
    float cnt = (float)totalCols;
    float sc[8], sh[8];
#pragma unroll
    for (int i = 0; i < 8; ++i) {
        int o = mbase + hi * 8 + i;
        float mu  = stats[2 * o] / cnt;
        float var = fmaxf(stats[2 * o + 1] / cnt - mu * mu, 0.f);
        float inv = rsqrtf(var + BN_EPS);
        sc[i] = g[o] * inv;
        sh[i] = bb[o] - mu * sc[i];
    }
    int T = totalCols / 16;
    int gw = blockIdx.x * 4 + wv, nw = gridDim.x * 4;
    for (int t = gw; t < T; t += nw) {
        v8f acc = {0.f,0.f,0.f,0.f,0.f,0.f,0.f,0.f};
#pragma unroll
        for (int ch = 0; ch < NC; ++ch) {
            BFrag Bf;
            const unsigned short* pb = fbf + (size_t)(t * 16 + lm) * Cfpad + ch * 32 + hi * 8;
            Bf.q[0] = *(const v4u*)pb;
            Bf.q[1] = *(const v4u*)(pb + 16);
            acc = __builtin_amdgcn_wmma_f32_16x16x32_bf16(false, A[ch].v, false, Bf.v,
                                                          (short)0, acc, false, false);
        }
        int col = t * 16 + lm;
        int b   = col / (NPTS * Kfac);
        int rem = col % (NPTS * Kfac);
        int n   = rem / Kfac;
        int nout = (Nout == 1) ? 0 : n;
#pragma unroll
        for (int i = 0; i < 8; ++i) {
            int o = mbase + hi * 8 + i;
            float y = acc[i] * sc[i] + sh[i];
            float l = y > 0.f ? y : 0.2f * y;
            atomicMaxF(&out[((size_t)b * O + o) * Nout + nout], l);
        }
    }
}

// ---------------------------------------------------------------- plane scatter-mean
__global__ void k_cells(const float* pts, int* cells) {
    int t = blockIdx.x * blockDim.x + threadIdx.x;
    if (t >= 3 * BATCH * NPTS) return;
    int n = t % NPTS;
    int b = (t / NPTS) % BATCH;
    int p = t / (NPTS * BATCH);
    int s0 = (p == 2) ? 1 : 0;       // xz:(0,2) xy:(0,1) yz:(1,2)
    int s1 = (p == 1) ? 1 : 2;
    float u = pts[((size_t)b * NPTS + n) * 3 + s0] / 1.101f + 0.5f;
    float w = pts[((size_t)b * NPTS + n) * 3 + s1] / 1.101f + 0.5f;
    u = fminf(fmaxf(u, 0.f), 1.f - 1e-5f);
    w = fminf(fmaxf(w, 0.f), 1.f - 1e-5f);
    int i0 = (int)(u * RESO), i1 = (int)(w * RESO);
    cells[t] = i0 + RESO * i1;
}

__global__ void k_count(const int* cells, float* cnt) {
    int t = blockIdx.x * blockDim.x + threadIdx.x;
    if (t >= 3 * BATCH * NPTS) return;
    int b = (t / NPTS) % BATCH;
    int p = t / (NPTS * BATCH);
    atomicAdd(&cnt[((size_t)p * BATCH + b) * (RESO * RESO) + cells[t]], 1.f);
}

__global__ void k_scatter(const int* cells, const float* pre,
                          const float* h0, const float* h1, const float* h2, const float* h3,
                          float* out) {
    long t = (long)blockIdx.x * blockDim.x + threadIdx.x;
    long total = 3L * BATCH * NPTS * 1024;
    if (t >= total) return;
    int  ch  = (int)(t % 1024);
    long rem = t / 1024;
    int  n   = (int)(rem % NPTS);
    int  b   = (int)((rem / NPTS) % BATCH);
    int  p   = (int)(rem / ((long)NPTS * BATCH));
    int  cell = cells[((size_t)p * BATCH + b) * NPTS + n];
    float v;
    if      (ch < 512) v = pre[(size_t)b * 512 + ch];
    else if (ch < 576) v = h0[((size_t)b * 64  + (ch - 512)) * NPTS + n];
    else if (ch < 640) v = h1[((size_t)b * 64  + (ch - 576)) * NPTS + n];
    else if (ch < 768) v = h2[((size_t)b * 128 + (ch - 640)) * NPTS + n];
    else               v = h3[((size_t)b * 256 + (ch - 768)) * NPTS + n];
    atomicAdd(&out[(((size_t)p * BATCH + b) * 1024 + ch) * (RESO * RESO) + cell], v);
}

__global__ void k_divide(float* out, const float* cnt) {
    long t = (long)blockIdx.x * blockDim.x + threadIdx.x;
    long total = 3L * BATCH * 1024 * (RESO * RESO);
    if (t >= total) return;
    int cell = (int)(t % (RESO * RESO));
    int b    = (int)((t / ((long)(RESO * RESO) * 1024)) % BATCH);
    int p    = (int)(t / ((long)(RESO * RESO) * 1024 * BATCH));
    float c = fmaxf(cnt[((size_t)p * BATCH + b) * (RESO * RESO) + cell], 1.f);
    out[t] /= c;
}

// ---------------------------------------------------------------- launcher
extern "C" void kernel_launch(void* const* d_in, const int* in_sizes, int n_in,
                              void* d_out, int out_size, void* d_ws, size_t ws_size,
                              hipStream_t stream) {
    (void)in_sizes; (void)n_in; (void)out_size; (void)ws_size;
    const float* pts = (const float*)d_in[0];
    const float* W[5]  = {(const float*)d_in[1], (const float*)d_in[4], (const float*)d_in[7],
                          (const float*)d_in[10], (const float*)d_in[13]};
    const float* G[5]  = {(const float*)d_in[2], (const float*)d_in[5], (const float*)d_in[8],
                          (const float*)d_in[11], (const float*)d_in[14]};
    const float* Bb[5] = {(const float*)d_in[3], (const float*)d_in[6], (const float*)d_in[9],
                          (const float*)d_in[12], (const float*)d_in[15]};

    char* base = (char*)d_ws;
    size_t off = 0;
    auto carve = [&](size_t bytes) -> void* {
        void* p = base + off;
        off += (bytes + 255) & ~(size_t)255;
        return p;
    };
    float*          xbuf  = (float*)carve((size_t)BATCH * 3 * NPTS * 4);
    float*          h0    = (float*)carve((size_t)BATCH * 64  * NPTS * 4);
    float*          h1    = (float*)carve((size_t)BATCH * 64  * NPTS * 4);
    float*          h2    = (float*)carve((size_t)BATCH * 128 * NPTS * 4);
    float*          h3    = (float*)carve((size_t)BATCH * 256 * NPTS * 4);
    unsigned short* hT    = (unsigned short*)carve((size_t)BATCH * NPTS * 128 * 2);
    float*          xx    = (float*)carve((size_t)BATCH * NPTS * 4);
    int*            idx   = (int*)carve((size_t)BATCH * NPTS * KNN_K * 4);
    unsigned short* wbf   = (unsigned short*)carve((size_t)512 * 512 * 2);
    unsigned short* fbf   = (unsigned short*)carve((size_t)BATCH * NPTS * KNN_K * 256 * 2);
    float*          stats = (float*)carve((size_t)512 * 2 * 4);
    int*            cells = (int*)carve((size_t)3 * BATCH * NPTS * 4);
    float*          cnt   = (float*)carve((size_t)3 * BATCH * RESO * RESO * 4);

    const int BN = BATCH * NPTS;
    k_transpose_pts<<<(BN + 255) / 256, 256, 0, stream>>>(pts, xbuf);

    struct Lay { const float* in; int Cin; int CpadK; int Cf; int Cfpad; int O; int nck; int nc; float* hout; };
    Lay L[4] = {
        { xbuf,   3,  32,   9,  32,  64, 1, 1, h0 },
        { h0,    64,  64, 128, 128,  64, 2, 4, h1 },
        { h1,    64,  64, 128, 128, 128, 2, 4, h2 },
        { h2,   128, 128, 256, 256, 256, 4, 8, h3 },
    };

    for (int l = 0; l < 4; ++l) {
        Lay& P = L[l];
        k_packT<<<(BN + 255) / 256, 256, 0, stream>>>(P.in, hT, xx, P.Cin, P.CpadK);
        int knnBlocks = BATCH * (NPTS / 16);
        switch (P.nck) {
            case 1: k_knn<1><<<knnBlocks, 128, 0, stream>>>(hT, xx, idx); break;
            case 2: k_knn<2><<<knnBlocks, 128, 0, stream>>>(hT, xx, idx); break;
            default: k_knn<4><<<knnBlocks, 128, 0, stream>>>(hT, xx, idx); break;
        }
        k_packW<<<(P.O * P.Cfpad + 255) / 256, 256, 0, stream>>>(W[l], wbf, P.O, P.Cf, P.Cfpad);
        long gt = (long)BATCH * NPTS * KNN_K * P.Cfpad;
        k_gather<<<(unsigned)((gt + 255) / 256), 256, 0, stream>>>(P.in, idx, fbf, P.Cin, P.Cfpad, l == 0 ? 1 : 0);
        k_fill<<<1, 256, 0, stream>>>(stats, 2L * P.O, 0.f);
        int totalCols = BATCH * NPTS * KNN_K;
        dim3 grid(128, P.O / 16);
        switch (P.nc) {
            case 1: k_conv_stats<1><<<grid, 128, 0, stream>>>(wbf, fbf, stats, totalCols); break;
            case 4: k_conv_stats<4><<<grid, 128, 0, stream>>>(wbf, fbf, stats, totalCols); break;
            default: k_conv_stats<8><<<grid, 128, 0, stream>>>(wbf, fbf, stats, totalCols); break;
        }
        long hn = (long)BATCH * P.O * NPTS;
        k_fill<<<(unsigned)((hn + 255) / 256), 256, 0, stream>>>(P.hout, hn, -3.0e38f);
        switch (P.nc) {
            case 1: k_conv_apply<1><<<grid, 128, 0, stream>>>(wbf, fbf, stats, G[l], Bb[l], P.hout, totalCols, KNN_K, P.O, NPTS); break;
            case 4: k_conv_apply<4><<<grid, 128, 0, stream>>>(wbf, fbf, stats, G[l], Bb[l], P.hout, totalCols, KNN_K, P.O, NPTS); break;
            default: k_conv_apply<8><<<grid, 128, 0, stream>>>(wbf, fbf, stats, G[l], Bb[l], P.hout, totalCols, KNN_K, P.O, NPTS); break;
        }
    }

    // ------- final conv over hcat [512] -> pre = max over N, stored at d_out[0..2048)
    long catN = (long)BN * 512;
    k_packcat<<<(unsigned)((catN + 255) / 256), 256, 0, stream>>>(h0, h1, h2, h3, fbf);
    k_packW<<<(512 * 512 + 255) / 256, 256, 0, stream>>>(W[4], wbf, 512, 512, 512);
    k_fill<<<1, 256, 0, stream>>>(stats, 1024L, 0.f);
    int fCols = BN;
    dim3 gridF(64, 32);
    k_conv_stats<16><<<gridF, 128, 0, stream>>>(wbf, fbf, stats, fCols);
    float* pre = (float*)d_out;
    k_fill<<<8, 256, 0, stream>>>(pre, 2048L, -3.0e38f);
    k_conv_apply<16><<<gridF, 128, 0, stream>>>(wbf, fbf, stats, G[4], Bb[4], pre, fCols, 1, 512, 1);

    // ------- plane scatter-mean into d_out[2048 ...)
    k_cells<<<(3 * BN + 255) / 256, 256, 0, stream>>>(pts, cells);
    float* planes = (float*)d_out + 2048;
    long pn = 3L * BATCH * 1024 * (RESO * RESO);
    k_fill<<<(unsigned)((pn + 255) / 256), 256, 0, stream>>>(planes, pn, 0.f);
    k_fill<<<(3 * BATCH * RESO * RESO + 255) / 256, 256, 0, stream>>>(cnt, 3L * BATCH * RESO * RESO, 0.f);
    k_count<<<(3 * BN + 255) / 256, 256, 0, stream>>>(cells, cnt);
    long sn = 3L * BATCH * NPTS * 1024;
    k_scatter<<<(unsigned)((sn + 255) / 256), 256, 0, stream>>>(cells, pre, h0, h1, h2, h3, planes);
    k_divide<<<(unsigned)((pn + 255) / 256), 256, 0, stream>>>(planes, cnt);
}
